// QuantizedCpuLinear_10634339025399
// MI455X (gfx1250) — 8-bit matrix operand order repaired and checked by reading, compiled, not run on hardware
//
#include <hip/hip_runtime.h>
#include <cstdint>


typedef __attribute__((ext_vector_type(8))) int v8i;

#define M_TOTAL 8192
#define N_TOTAL 4096
#define K_TOTAL 4096
#define KT      (K_TOTAL / 64)

__global__ __launch_bounds__(256) void quantize_pack_A(
    const float* __restrict__ x, const float* __restrict__ input_scale,
    uint32_t* __restrict__ Apk)
{
    const size_t d    = (size_t)blockIdx.x * 256u + threadIdx.x;
    const int    v    = (int)(d & 7u);
    const int    lane = (int)((d >> 3) & 31u);
    const size_t frag = d >> 8;
    const int    kt   = (int)(frag & (KT - 1));
    const size_t mt   = frag >> 6;
    const int    h    = lane >> 4;
    const size_t m    = mt * 16 + (lane & 15);
    const int    kb   = kt * 64 + ((v >> 1) << 4) + (h << 3) + ((v & 1) << 2);

    const float s = input_scale[0];
    const float4 xv = *(const float4*)(x + m * (size_t)K_TOTAL + kb);

    int q0 = __float2int_rn(fminf(fmaxf(xv.x / s, -128.f), 127.f));
    int q1 = __float2int_rn(fminf(fmaxf(xv.y / s, -128.f), 127.f));
    int q2 = __float2int_rn(fminf(fmaxf(xv.z / s, -128.f), 127.f));
    int q3 = __float2int_rn(fminf(fmaxf(xv.w / s, -128.f), 127.f));

    Apk[d] = (uint32_t)(q0 & 0xff) | ((uint32_t)(q1 & 0xff) << 8) |
             ((uint32_t)(q2 & 0xff) << 16) | ((uint32_t)(q3 & 0xff) << 24);
}

__global__ __launch_bounds__(256) void pack_B(
    const int* __restrict__ w, uint32_t* __restrict__ Bpk)
{
    const size_t d    = (size_t)blockIdx.x * 256u + threadIdx.x;
    const int    v    = (int)(d & 7u);
    const int    lane = (int)((d >> 3) & 31u);
    const size_t frag = d >> 8;
    const int    kt   = (int)(frag & (KT - 1));
    const size_t nt   = frag >> 6;
    const int    h    = lane >> 4;
    const size_t n    = nt * 16 + (lane & 15);
    const int    kb   = kt * 64 + ((v >> 1) << 4) + (h << 3) + ((v & 1) << 2);

    const int4 wv = *(const int4*)(w + n * (size_t)K_TOTAL + kb);

    Bpk[d] = (uint32_t)(wv.x & 0xff) | ((uint32_t)(wv.y & 0xff) << 8) |
             ((uint32_t)(wv.z & 0xff) << 16) | ((uint32_t)(wv.w & 0xff) << 24);
}

__global__ __launch_bounds__(256, 1) void gemm_iu8(
    const v8i* __restrict__ Apk, const v8i* __restrict__ Bpk,
    const float* __restrict__ scale, const float* __restrict__ input_scale,
    float* __restrict__ out)
{
    const int lane = threadIdx.x & 31;
    const int wave = threadIdx.x >> 5;
    const int wm   = wave & 1;
    const int wn   = wave >> 1;

    const int block_m = blockIdx.x * 128 + wm * 64;
    const int block_n = blockIdx.y * 128 + wn * 32;
    const int mt0 = block_m >> 4;
    const int nt0 = block_n >> 4;

    const v8i zero = {0, 0, 0, 0, 0, 0, 0, 0};
    v8i acc[4][2];
#pragma unroll
    for (int i = 0; i < 4; ++i)
#pragma unroll
        for (int j = 0; j < 2; ++j) acc[i][j] = zero;

    for (int kt = 0; kt < KT; ++kt) {
        if (kt + 1 < KT) {
            __builtin_prefetch(&Apk[((size_t)mt0 * KT + kt + 1) * 32 + lane], 0, 1);
            __builtin_prefetch(&Bpk[((size_t)nt0 * KT + kt + 1) * 32 + lane], 0, 1);
        }

        v8i a[4], b[2];
#pragma unroll
        for (int i = 0; i < 4; ++i)
            a[i] = Apk[((size_t)(mt0 + i) * KT + kt) * 32 + lane];
#pragma unroll
        for (int j = 0; j < 2; ++j)
            b[j] = Bpk[((size_t)(nt0 + j) * KT + kt) * 32 + lane];

#pragma unroll
        for (int i = 0; i < 4; ++i)
#pragma unroll
            for (int j = 0; j < 2; ++j)
                acc[i][j] = __builtin_amdgcn_wmma_i32_16x16x64_iu8(
                     true, a[i],  true, b[j],
                    acc[i][j],  false,  false);
    }

    const float s = input_scale[0] * scale[0];

    const int h    = lane >> 4;
    const int n_in = lane & 15;
#pragma unroll
    for (int i = 0; i < 4; ++i) {
#pragma unroll
        for (int j = 0; j < 2; ++j) {
            const size_t col = (size_t)(block_n + j * 16 + n_in);
#pragma unroll
            for (int g = 0; g < 8; ++g) {
                const size_t row = (size_t)(block_m + i * 16 + h * 8 + g);
                out[row * N_TOTAL + col] = (float)acc[i][j][g] * s;
            }
        }
    }
}

extern "C" void kernel_launch(void* const* d_in, const int* in_sizes, int n_in,
                              void* d_out, int out_size, void* d_ws, size_t ws_size,
                              hipStream_t stream)
{
    const float* x           = (const float*)d_in[0];
    const int*   weight      = (const int*)d_in[1];
    const float* scale       = (const float*)d_in[2];
    const float* input_scale = (const float*)d_in[3];
    float*       out         = (float*)d_out;

    uint8_t*  ws  = (uint8_t*)d_ws;
    uint32_t* Apk = (uint32_t*)ws;
    uint32_t* Bpk = (uint32_t*)(ws + (size_t)M_TOTAL * K_TOTAL);

    {
        const size_t dwords = (size_t)M_TOTAL * K_TOTAL / 4;
        quantize_pack_A<<<dim3((unsigned)(dwords / 256)), dim3(256), 0, stream>>>(
            x, input_scale, Apk);
    }
    {
        const size_t dwords = (size_t)N_TOTAL * K_TOTAL / 4;
        pack_B<<<dim3((unsigned)(dwords / 256)), dim3(256), 0, stream>>>(
            weight, Bpk);
    }
    {
        dim3 grid(M_TOTAL / 128, N_TOTAL / 128);
        gemm_iu8<<<grid, dim3(256), 0, stream>>>(
            (const v8i*)Apk, (const v8i*)Bpk, scale, input_scale, out);
    }
}
